// KPFCNN_1932735283424
// MI455X (gfx1250) — matrix operand order repaired and checked by reading, not compiled, not run on hardware
//
#include <hip/hip_runtime.h>

// ---------------------------------------------------------------------------
// KPFCNN on MI455X (gfx1250, wave32, WMMA f16 -> f32 accumulate).
//  * stage-1 neighbor gathers: 16-byte global_load_b128 into wave-private LDS
//  * layer 1 (Cin=1): shfl-butterfly VALU stage-1 (no divergent gather)
//  * WMMA B operands: pre-packed fragment-native f16 weights
//  * detection scores: precomputed row-sums kill per-neighbor wave reductions
// ---------------------------------------------------------------------------

#define N0C 60000
#define N1C 15000
#define N2C 3750
#define KNB 32
#define PKP 15
#define R0f 0.5f

typedef __attribute__((ext_vector_type(16))) _Float16 v16h;
typedef __attribute__((ext_vector_type(8)))  _Float16 v8h;
typedef __attribute__((ext_vector_type(8)))  float    v8f;

__device__ __forceinline__ float leakyf(float x) { return x > 0.f ? x : 0.1f * x; }

__device__ __forceinline__ float waveMaxF(float v) {
#pragma unroll
  for (int s = 16; s > 0; s >>= 1) v = fmaxf(v, __shfl_xor(v, s, 32));
  return v;
}
__device__ __forceinline__ float waveSumF(float v) {
#pragma unroll
  for (int s = 16; s > 0; s >>= 1) v += __shfl_xor(v, s, 32);
  return v;
}

// ---------------------------------------------------------------------------
// Weight packing: f32 [P][CIN][COUT] -> f16 fragment-native layout
//   flat K index r = c*16 + p (c-major, matches fk LDS tile layout)
// ---------------------------------------------------------------------------
template <int CIN, int CINP, int COUT>
__global__ void __launch_bounds__(256)
pack_kpw_kernel(const float* __restrict__ W, _Float16* __restrict__ Wp) {
  constexpr int K2 = 16 * CINP;
  const int i = blockIdx.x * 256 + threadIdx.x;
  if (i >= K2 * COUT) return;
  const int within = i & 31;
  const int col    = (i >> 5) % COUT;
  const int blk    = i / (32 * COUT);
  const int r      = blk * 32 + within;
  const int p      = r & 15;
  const int c      = r >> 4;
  float v = 0.f;
  if (p < PKP && c < CIN) v = W[((long)p * CIN + c) * COUT + col];
  Wp[((long)blk * COUT + col) * 32 + within] = (_Float16)v;
}

template <int KD, int COUT>
__global__ void __launch_bounds__(256)
pack_uw_kernel(const float* __restrict__ W, _Float16* __restrict__ Wp) {
  const int i = blockIdx.x * 256 + threadIdx.x;
  if (i >= KD * COUT) return;
  const int within = i & 31;
  const int col    = (i >> 5) % COUT;
  const int blk    = i / (32 * COUT);
  Wp[((long)blk * COUT + col) * 32 + within] =
      (_Float16)W[((long)blk * 32 + within) * COUT + col];
}

__global__ void __launch_bounds__(256)
pack_f16_kernel(const float* __restrict__ x, _Float16* __restrict__ y, long n) {
  const long i = (long)blockIdx.x * 256 + threadIdx.x;
  if (i < n) y[i] = (_Float16)x[i];
}

// ---------------------------------------------------------------------------
// Fused rigid KPConv layer (block = 256 thr = 8 waves, 16 query pts / block).
// ---------------------------------------------------------------------------
template <int CIN, int CINP, int COUT>
__global__ void __launch_bounds__(256)
kpconv_kernel(int Nq, int Ns,
              const float*    __restrict__ qpts,
              const float*    __restrict__ spts,
              const int*      __restrict__ neigh,
              const _Float16* __restrict__ featsh,
              const float*    __restrict__ kpts,
              const _Float16* __restrict__ Wp,
              float extent,
              float*    __restrict__ out,
              _Float16* __restrict__ outh)
{
  constexpr int K2     = 16 * CINP;
  constexpr int CTILES = CINP / 16;
  constexpr int NTILES = COUT / 16;
  constexpr bool TINY  = (CIN < 16);

  extern __shared__ unsigned char smem_raw[];
  float*    sdiff = (float*)smem_raw;                       // [16][32][3]
  int*      snidx = (int*)(sdiff + 16 * KNB * 3);           // [16][32]
  _Float16* sfk   = (_Float16*)(snidx + 16 * KNB);          // [16][K2] c-major
  float*    sFv   = (float*)(sfk + 16 * K2);                // TINY: [16][32]
  _Float16* sB    = (_Float16*)(sfk + 16 * K2);             // else: [8][32][CIN]

  const int tid   = threadIdx.x;
  const int lane  = tid & 31;
  const int wid   = tid >> 5;
  const int qbase = blockIdx.x * 16;

  // ---- Phase 0: gather neighbor indices + relative coords (+ feats if TINY) ----
  for (int pair = tid; pair < 16 * KNB; pair += 256) {
    const int q = pair >> 5, k = pair & 31;
    const int gq = qbase + q;
    float dx = 1e9f, dy = 1e9f, dz = 1e9f, fv = 0.f;
    int idx = 0;
    if (gq < Nq) {
      const int raw = neigh[(long)gq * KNB + k];
      const float qx = qpts[(long)gq * 3 + 0];
      const float qy = qpts[(long)gq * 3 + 1];
      const float qz = qpts[(long)gq * 3 + 2];
      if (raw >= 0 && raw < Ns) {
        dx = spts[(long)raw * 3 + 0] - qx;
        dy = spts[(long)raw * 3 + 1] - qy;
        dz = spts[(long)raw * 3 + 2] - qz;
        idx = raw;
        if (TINY) fv = (float)featsh[raw];
      } else {
        dx = 1e6f - qx; dy = 1e6f - qy; dz = 1e6f - qz;
      }
    }
    sdiff[(q * KNB + k) * 3 + 0] = dx;
    sdiff[(q * KNB + k) * 3 + 1] = dy;
    sdiff[(q * KNB + k) * 3 + 2] = dz;
    snidx[q * KNB + k] = idx;
    if (TINY) sFv[q * KNB + k] = fv;
  }
  if (TINY) {
    v8h z = {};
    for (int i = tid; i < 16 * K2 / 8; i += 256) ((v8h*)sfk)[i] = z;
  }
  __syncthreads();

  // ---- Stage 1: 2 queries per wave ----
  const int p = lane & 15;
  float kpx = 0.f, kpy = 0.f, kpz = 0.f;
  if (p < PKP) {
    kpx = kpts[p * 3 + 0] * extent;
    kpy = kpts[p * 3 + 1] * extent;
    kpz = kpts[p * 3 + 2] * extent;
  }
  const float pvalid  = (p < PKP) ? 1.f : 0.f;
  const float inv_ext = 1.0f / extent;
  const int aoff = (lane < 16) ? 0 : 8;
  const int boff = (lane < 16) ? 0 : 16;
  const int n    = lane & 15;
  const int mb   = (lane >> 4) * 8;
  _Float16* sBw  = sB + wid * 32 * CIN;

  for (int q = wid * 2; q < wid * 2 + 2; ++q) {
    float infl[16];
#pragma unroll
    for (int i = 0; i < 16; ++i) {
      const int k = (i < 8) ? (aoff + i) : (16 + aoff + (i - 8));
      const float ddx = sdiff[(q * KNB + k) * 3 + 0] - kpx;
      const float ddy = sdiff[(q * KNB + k) * 3 + 1] - kpy;
      const float ddz = sdiff[(q * KNB + k) * 3 + 2] - kpz;
      const float dist = sqrtf(ddx * ddx + ddy * ddy + ddz * ddz);
      infl[i] = fmaxf(0.f, 1.f - dist * inv_ext) * pvalid;
    }

    if constexpr (TINY) {
      float s = 0.f;
#pragma unroll
      for (int i = 0; i < 16; ++i) {
        const int k = (i < 8) ? (aoff + i) : (16 + aoff + (i - 8));
        s += infl[i] * sFv[q * KNB + k];
      }
      s += __shfl_xor(s, 16, 32);
      if (lane < 16) sfk[q * K2 + p] = (_Float16)s;
    } else {
      v16h a;
#pragma unroll
      for (int i = 0; i < 16; ++i) a[i] = (_Float16)infl[i];

      {
        const int ridx = snidx[q * KNB + lane];
        const _Float16* srow = featsh + (long)ridx * CIN;
#pragma unroll
        for (int j = 0; j < CIN / 8; ++j)
          *(v8h*)(sBw + lane * CIN + j * 8) = *(const v8h*)(srow + j * 8);
      }

#pragma unroll
      for (int ct = 0; ct < CTILES; ++ct) {
        const int c = ct * 16 + n;
        v16h b;
#pragma unroll
        for (int i = 0; i < 16; ++i) b[i] = sBw[((i < 8) ? (aoff + i) : (16 + aoff + (i - 8))) * CIN + c];
        v8f acc = {};
        acc = __builtin_amdgcn_wmma_f32_16x16x32_f16(false, a, false, b,
                                                     (short)0, acc, false, false);
        v8h pk;
#pragma unroll
        for (int r = 0; r < 8; ++r) pk[r] = (_Float16)acc[r];
        *(v8h*)(sfk + q * K2 + c * 16 + mb) = pk;
      }
    }
  }
  __syncthreads();

  // ---- Stage 2: [16 x K2] @ packed Wp ----
  for (int nt = wid; nt < NTILES; nt += 8) {
    const int nb  = nt * 16;
    const int col = nb + n;
    const int m   = lane & 15;
    v8f acc = {};
    for (int kb = 0; kb < K2; kb += 32) {
      const _Float16* arow = sfk + m * K2 + kb;
      const v8h a0 = *(const v8h*)(arow + aoff);
      const v8h a1 = *(const v8h*)(arow + 16 + aoff);
      const v16h a = __builtin_shufflevector(a0, a1, 0,1,2,3,4,5,6,7,
                                                     8,9,10,11,12,13,14,15);
      const _Float16* wrow = Wp + ((long)(kb >> 5) * COUT + col) * 32 + aoff;
      const v8h b0 = *(const v8h*)wrow;
      const v8h b1 = *(const v8h*)(wrow + 16);
      const v16h b = __builtin_shufflevector(b0, b1, 0,1,2,3,4,5,6,7,
                                                     8,9,10,11,12,13,14,15);
      acc = __builtin_amdgcn_wmma_f32_16x16x32_f16(false, a, false, b,
                                                   (short)0, acc, false, false);
    }
#pragma unroll
    for (int r = 0; r < 8; ++r) {
      const int gq = qbase + mb + r;
      if (gq < Nq) {
        const float v = leakyf(acc[r]);
        out [(long)gq * COUT + nb + n] = v;
        outh[(long)gq * COUT + nb + n] = (_Float16)v;
      }
    }
  }
}

// ---------------------------------------------------------------------------
// Decoder unary: out[n] = act( concat(srcA[up[n]], srcB[n]) @ Wu ).
// ---------------------------------------------------------------------------
template <int K1, int K2C, int COUT, bool RELU>
__global__ void __launch_bounds__(128)
unary_kernel(int N,
             const _Float16* __restrict__ srcAh,
             const int*      __restrict__ up,
             const _Float16* __restrict__ srcBh,
             const _Float16* __restrict__ Wp,
             float*    __restrict__ out,
             _Float16* __restrict__ outh)
{
  constexpr int KD = K1 + K2C;
  const int lane  = threadIdx.x & 31;
  const int wid   = threadIdx.x >> 5;
  const int qbase = blockIdx.x * 16;
  const int m     = lane & 15;
  const int gq    = qbase + m;
  const int cgq   = (gq < N) ? gq : (N - 1);
  const int ia    = up[cgq];
  const int aoff  = (lane < 16) ? 0 : 8;
  const int boff  = (lane < 16) ? 0 : 16;
  const _Float16* rowA = srcAh + (long)ia * K1;
  const _Float16* rowB = srcBh + (long)cgq * K2C;

  for (int nt = wid; nt < COUT / 16; nt += 4) {
    const int col = nt * 16 + (lane & 15);
    v8f acc = {};
    for (int kb = 0; kb < KD; kb += 32) {
      const int r0 = kb + aoff;
      const int r1 = kb + 16 + aoff;
      const _Float16* s0 = (r0 < K1) ? (rowA + r0) : (rowB + (r0 - K1));
      const _Float16* s1 = (r1 < K1) ? (rowA + r1) : (rowB + (r1 - K1));
      const v8h a0 = *(const v8h*)s0;
      const v8h a1 = *(const v8h*)s1;
      const v16h a = __builtin_shufflevector(a0, a1, 0,1,2,3,4,5,6,7,
                                                     8,9,10,11,12,13,14,15);
      const _Float16* wrow = Wp + ((long)(kb >> 5) * COUT + col) * 32 + aoff;
      const v8h b0 = *(const v8h*)wrow;
      const v8h b1 = *(const v8h*)(wrow + 16);
      const v16h b = __builtin_shufflevector(b0, b1, 0,1,2,3,4,5,6,7,
                                                     8,9,10,11,12,13,14,15);
      acc = __builtin_amdgcn_wmma_f32_16x16x32_f16(false, a, false, b,
                                                   (short)0, acc, false, false);
    }
    const int mbv = (lane >> 4) * 8;
    const int n   = nt * 16 + (lane & 15);
#pragma unroll
    for (int r = 0; r < 8; ++r) {
      const int g = qbase + mbv + r;
      if (g < N) {
        float v = acc[r];
        if (RELU) v = leakyf(v);
        out [(long)g * COUT + n] = v;
        outh[(long)g * COUT + n] = (_Float16)v;
      }
    }
  }
}

// ---------------------------------------------------------------------------
// Global max (order-preserving uint encoding + atomicMax).
// ---------------------------------------------------------------------------
__global__ void init_gmax_kernel(unsigned* p) {
  if (threadIdx.x == 0 && blockIdx.x == 0) *p = 0u;
}

__global__ void __launch_bounds__(256)
reduce_max_kernel(const float* __restrict__ x, long n, unsigned* __restrict__ outp) {
  float m = -INFINITY;
  for (long i = (long)blockIdx.x * blockDim.x + threadIdx.x; i < n;
       i += (long)gridDim.x * blockDim.x)
    m = fmaxf(m, x[i]);
  m = waveMaxF(m);
  __shared__ float sm[8];
  const int wid = threadIdx.x >> 5, lane = threadIdx.x & 31;
  if (lane == 0) sm[wid] = m;
  __syncthreads();
  if (threadIdx.x == 0) {
    float mm = sm[0];
    for (int i = 1; i < 8; ++i) mm = fmaxf(mm, sm[i]);
    unsigned u = __float_as_uint(mm);
    u = (u & 0x80000000u) ? ~u : (u | 0x80000000u);
    atomicMax(outp, u);
  }
}

// ---------------------------------------------------------------------------
// Per-point channel sums: rs[n] = sum_c f[n,c]  (vectorized, one thread/point).
// Used by the scores kernel: sum_c nf[k] = inv * rs[idx_k], and inv != 0,
// so the "neighbor nonzero" test needs no per-neighbor wave reduction.
// ---------------------------------------------------------------------------
__global__ void __launch_bounds__(256)
rowsum_kernel(int N, const float* __restrict__ f, float* __restrict__ rs) {
  const int nid = blockIdx.x * 256 + threadIdx.x;
  if (nid >= N) return;
  const v8f* row = (const v8f*)(f + (long)nid * 32);
  float s = 0.f;
#pragma unroll
  for (int j = 0; j < 4; ++j) {
    const v8f v = row[j];
#pragma unroll
    for (int e = 0; e < 8; ++e) s += v[e];
  }
  rs[nid] = s;
}

// ---------------------------------------------------------------------------
// Detection scores + L2 normalization. One wave per point; lane == channel.
// Neighbor-count pass uses lane = k (one butterfly); main loop has zero
// cross-lane ops (neighbor index is a wave-uniform scalar load).
// ---------------------------------------------------------------------------
__global__ void __launch_bounds__(256)
scores_norm_kernel(int N,
                   const float* __restrict__ f,       // [N,32]
                   const float* __restrict__ rs,      // [N] row sums
                   const int*   __restrict__ neigh,   // [N,32]
                   const unsigned* __restrict__ gmax_enc,
                   float* __restrict__ out_fnorm,     // [N,32]
                   float* __restrict__ out_scores)    // [N]
{
  const int lane = threadIdx.x & 31;
  const int wid  = threadIdx.x >> 5;
  const unsigned ue = *gmax_enc;
  const float g = __uint_as_float((ue & 0x80000000u) ? (ue & 0x7fffffffu) : ~ue);
  const float inv = 1.f / (g + 1e-6f);

  for (long n = (long)blockIdx.x * 8 + wid; n < N; n += (long)gridDim.x * 8) {
    // ---- pass 1 (lane = k): neighbor count, single butterfly ----
    const int myk  = neigh[n * KNB + lane];
    const bool okk = (myk >= 0 && myk < N);
    const float flag = (okk && rs[okk ? myk : 0] != 0.f) ? 1.f : 0.f;
    const float cnt  = waveSumF(flag);
    const float nnum = fmaxf(cnt, 1.f);

    // ---- pass 2 (lane = channel): per-channel sums/maxes over neighbors ----
    const float fc = f[n * 32 + lane];
    const float fn = fc * inv;
    float ssum = 0.f, smax = -INFINITY;
    for (int k = 0; k < KNB; ++k) {
      const int idx = neigh[n * KNB + k];        // wave-uniform -> scalar load
      float v = 0.f;
      if (idx >= 0 && idx < N) v = f[(long)idx * 32 + lane] * inv;
      ssum += v;
      smax = fmaxf(smax, v);
    }
    const float meanc = ssum / nnum;
    const float xdm   = fn - meanc;
    const float lms   = (xdm > 20.f) ? xdm : log1pf(expf(xdm));  // softplus
    const float dmax  = waveMaxF(fn);
    const float all   = lms * fn / (1e-6f + dmax);
    const float sc    = waveMaxF(all);
    float det = (fn == smax) ? 1.f : 0.f;
    det = waveMaxF(det);

    const float nrm = sqrtf(waveSumF(fc * fc));
    out_fnorm[n * 32 + lane] = fc / fmaxf(nrm, 1e-12f);
    if (lane == 0) out_scores[n] = sc * det;
  }
}

// ---------------------------------------------------------------------------
// Orchestration.
// ---------------------------------------------------------------------------
extern "C" void kernel_launch(void* const* d_in, const int* in_sizes, int n_in,
                              void* d_out, int out_size, void* d_ws, size_t ws_size,
                              hipStream_t stream)
{
  const float* features = (const float*)d_in[0];
  const float* pts0 = (const float*)d_in[1];
  const float* pts1 = (const float*)d_in[2];
  const float* pts2 = (const float*)d_in[3];
  const int*   nb0  = (const int*)d_in[4];
  const int*   nb1  = (const int*)d_in[5];
  const int*   nb2  = (const int*)d_in[6];
  const int*   pl0  = (const int*)d_in[7];
  const int*   pl1  = (const int*)d_in[8];
  const int*   up0  = (const int*)d_in[9];
  const int*   up1  = (const int*)d_in[10];
  const float* kpts = (const float*)d_in[11];
  const float* W1   = (const float*)d_in[12];
  const float* W2   = (const float*)d_in[13];
  const float* W3   = (const float*)d_in[14];
  const float* W4   = (const float*)d_in[15];
  const float* W5   = (const float*)d_in[16];
  const float* Wu1  = (const float*)d_in[17];
  const float* Wu2  = (const float*)d_in[18];

  float* f1  = (float*)d_ws;                    // [N0,64]  skip0
  float* f3  = f1  + (size_t)N0C * 64;          // [N1,128] skip1
  float* fu2 = f3  + (size_t)N1C * 128;         // [N0,32]  final features
  float* f2  = fu2 + (size_t)N0C * 32;          // [N1,64]
  float* f4  = f2  + (size_t)N1C * 64;          // [N2,128]
  float* f5  = f4  + (size_t)N2C * 128;         // [N2,256]
  float* fu1 = f5  + (size_t)N2C * 256;         // [N1,128]
  _Float16* f0h  = (_Float16*)(fu1 + (size_t)N1C * 128);  // [N0,1]
  _Float16* f1h  = f0h  + (size_t)N0C * 1;
  _Float16* f2h  = f1h  + (size_t)N0C * 64;
  _Float16* f3h  = f2h  + (size_t)N1C * 64;
  _Float16* f4h  = f3h  + (size_t)N1C * 128;
  _Float16* f5h  = f4h  + (size_t)N2C * 128;
  _Float16* fu1h = f5h  + (size_t)N2C * 256;
  _Float16* fu2h = fu1h + (size_t)N1C * 128;
  _Float16* W1p  = fu2h + (size_t)N0C * 32;
  _Float16* W2p  = W1p + (size_t)256  * 64;
  _Float16* W3p  = W2p + (size_t)1024 * 64;
  _Float16* W4p  = W3p + (size_t)1024 * 128;
  _Float16* W5p  = W4p + (size_t)2048 * 128;
  _Float16* Wu1p = W5p + (size_t)2048 * 256;
  _Float16* Wu2p = Wu1p + (size_t)384 * 128;
  unsigned* gmax = (unsigned*)(Wu2p + (size_t)192 * 32);
  float*    rs   = (float*)(gmax + 1);          // [N0] row sums

  float* out_fnorm  = (float*)d_out;
  float* out_scores = out_fnorm + (size_t)N0C * 32;

  const int smemL1  = 6144 + 2048 + 16 * 256  * 2 + 2048;              // 18432
  const int smem64  = 6144 + 2048 + 16 * 1024 * 2 + 8 * 32 * 64 * 2;   // 73728
  const int smem128 = 6144 + 2048 + 16 * 2048 * 2 + 8 * 32 * 128 * 2;  // 139264

  pack_kpw_kernel<1, 16, 64>    <<<(256  * 64  + 255) / 256, 256, 0, stream>>>(W1, W1p);
  pack_kpw_kernel<64, 64, 64>   <<<(1024 * 64  + 255) / 256, 256, 0, stream>>>(W2, W2p);
  pack_kpw_kernel<64, 64, 128>  <<<(1024 * 128 + 255) / 256, 256, 0, stream>>>(W3, W3p);
  pack_kpw_kernel<128, 128, 128><<<(2048 * 128 + 255) / 256, 256, 0, stream>>>(W4, W4p);
  pack_kpw_kernel<128, 128, 256><<<(2048 * 256 + 255) / 256, 256, 0, stream>>>(W5, W5p);
  pack_uw_kernel<384, 128><<<(384 * 128 + 255) / 256, 256, 0, stream>>>(Wu1, Wu1p);
  pack_uw_kernel<192, 32> <<<(192 * 32  + 255) / 256, 256, 0, stream>>>(Wu2, Wu2p);
  pack_f16_kernel<<<(N0C + 255) / 256, 256, 0, stream>>>(features, f0h, (long)N0C);

  kpconv_kernel<1, 16, 64><<<(N0C + 15) / 16, 256, smemL1, stream>>>(
      N0C, N0C, pts0, pts0, nb0, f0h, kpts, W1p, R0f, f1, f1h);
  kpconv_kernel<64, 64, 64><<<(N1C + 15) / 16, 256, smem64, stream>>>(
      N1C, N0C, pts1, pts0, pl0, f1h, kpts, W2p, R0f, f2, f2h);
  kpconv_kernel<64, 64, 128><<<(N1C + 15) / 16, 256, smem64, stream>>>(
      N1C, N1C, pts1, pts1, nb1, f2h, kpts, W3p, 2.f * R0f, f3, f3h);
  kpconv_kernel<128, 128, 128><<<(N2C + 15) / 16, 256, smem128, stream>>>(
      N2C, N1C, pts2, pts1, pl1, f3h, kpts, W4p, 2.f * R0f, f4, f4h);
  kpconv_kernel<128, 128, 256><<<(N2C + 15) / 16, 256, smem128, stream>>>(
      N2C, N2C, pts2, pts2, nb2, f4h, kpts, W5p, 4.f * R0f, f5, f5h);

  unary_kernel<256, 128, 128, true><<<(N1C + 15) / 16, 128, 0, stream>>>(
      N1C, f5h, up1, f3h, Wu1p, fu1, fu1h);
  unary_kernel<128, 64, 32, false><<<(N0C + 15) / 16, 128, 0, stream>>>(
      N0C, fu1h, up0, f1h, Wu2p, fu2, fu2h);

  init_gmax_kernel<<<1, 32, 0, stream>>>(gmax);
  reduce_max_kernel<<<512, 256, 0, stream>>>(fu2, (long)N0C * 32, gmax);
  rowsum_kernel<<<(N0C + 255) / 256, 256, 0, stream>>>(N0C, fu2, rs);
  scores_norm_kernel<<<2048, 256, 0, stream>>>(N0C, fu2, rs, nb0, gmax,
                                               out_fnorm, out_scores);
}
